// DGCNNEncoder_37263136260292
// MI455X (gfx1250) — matrix operand order repaired and checked by reading, not compiled, not run on hardware
//
#include <hip/hip_runtime.h>
#include <hip/hip_bf16.h>
#include <stdint.h>

typedef __attribute__((ext_vector_type(16))) _Float16 v16h;
typedef __attribute__((ext_vector_type(8)))  _Float16 v8h;
typedef __attribute__((ext_vector_type(8)))  float    v8f;

#define NPTS  2048
#define KNN   32
#define BATCH 2
#define EPSN  1e-5f
#define SLOPE 0.2f

// ---------- WMMA helpers (CDNA5 v_wmma_f32_16x16x32_f16, wave32) ----------
// A operand (16x32 f16, row-major source): lanes 0-15 & 16-31 both map row = lane&15;
// low lanes carry K {0..7, 16..23}, high lanes K {8..15, 24..31}.
__device__ __forceinline__ v16h ldA(const _Float16* row, int kk, int lane) {
  int koff = (lane & 16) ? 8 : 0;
  v8h lo = *(const v8h*)(row + kk + koff);
  v8h hi = *(const v8h*)(row + kk + 16 + koff);
  v16h a;
#pragma unroll
  for (int i = 0; i < 8; ++i) { a[i] = lo[i]; a[i + 8] = hi[i]; }
  return a;
}
// B operand (32x16 f16): lane n holds column n; low lanes K {0..7, 16..23}, high lanes K {8..15, 24..31},
// from the (row-major [N, C]) source row.
__device__ __forceinline__ v16h ldB(const _Float16* col, int kk, int lane) {
  int koff = (lane & 16) ? 8 : 0;
  v8h x0 = *(const v8h*)(col + kk + koff);
  v8h x1 = *(const v8h*)(col + kk + 16 + koff);
  v16h b;
#pragma unroll
  for (int i = 0; i < 8; ++i) { b[i] = x0[i]; b[i + 8] = x1[i]; }
  return b;
}
__device__ __forceinline__ v8f wmma16(v16h a, v16h b, v8f c) {
  return __builtin_amdgcn_wmma_f32_16x16x32_f16(false, a, false, b, (short)0, c, false, false);
}
__device__ __forceinline__ v8f zero8() { v8f z = {}; return z; }

// ---------- small utility kernels ----------
__global__ void k_zero_f32(float* p, int n) {
  int t = blockIdx.x * blockDim.x + threadIdx.x;
  for (int i = t; i < n; i += gridDim.x * blockDim.x) p[i] = 0.f;
}
__global__ void k_zero_f16(_Float16* p, int n) {
  int t = blockIdx.x * blockDim.x + threadIdx.x;
  for (int i = t; i < n; i += gridDim.x * blockDim.x) p[i] = (_Float16)0.f;
}
__global__ void k_cvt_f32_f16(const float* in, _Float16* out, int n) {
  int t = blockIdx.x * blockDim.x + threadIdx.x;
  if (t < n) out[t] = (_Float16)in[t];
}
// x[B,N,3] -> h16[B,N,128] zero padded
__global__ void k_convert_x(const float* x, _Float16* h16) {
  int t = blockIdx.x * blockDim.x + threadIdx.x;   // B*N*128
  int c = t & 127, pn = t >> 7;
  float v = (c < 3) ? x[pn * 3 + c] : 0.f;
  h16[t] = (_Float16)v;
}
// squared norms per point from padded f16 features
__global__ void k_sq(const _Float16* h16, float* sq) {
  int t = blockIdx.x * blockDim.x + threadIdx.x;   // B*N
  const _Float16* r = h16 + ((size_t)t << 7);
  float s = 0.f;
  for (int c = 0; c < 128; ++c) { float v = (float)r[c]; s += v * v; }
  sq[t] = s;
}
// W[O,2C] -> W1h[O,128]=W[:, :C], Wmh[O,128]=W[:, C:]-W[:, :C], f16 zero-padded
__global__ void k_prep_w(const float* W, _Float16* W1h, _Float16* Wmh, int Cin) {
  int t = blockIdx.x * blockDim.x + threadIdx.x;   // O*128
  int c = t & 127, o = t >> 7;
  float w1 = 0.f, wm = 0.f;
  if (c < Cin) { w1 = W[o * 2 * Cin + c]; wm = W[o * 2 * Cin + Cin + c] - w1; }
  W1h[t] = (_Float16)w1;
  Wmh[t] = (_Float16)wm;
}

// ---------- pairwise distance GEMM, 16x64 tile/wave: D = |i|^2+|j|^2-2 i.j ----------
__global__ void k_dist4(const _Float16* h16, const float* sq, float* D) {
  int lane = threadIdx.x & 31;
  int wave = blockIdx.x * (blockDim.x >> 5) + (threadIdx.x >> 5); // B*128*32 waves
  int tj4 = wave & 31, ti = (wave >> 5) & 127, b = wave >> 12;
  int m = lane & 15;
  const _Float16* ab = h16 + ((size_t)(b * NPTS + ti * 16 + m) << 7);
  const _Float16* bb[4];
#pragma unroll
  for (int u = 0; u < 4; ++u)
    bb[u] = h16 + ((size_t)(b * NPTS + tj4 * 64 + u * 16 + m) << 7);
  v8f acc[4];
#pragma unroll
  for (int u = 0; u < 4; ++u) acc[u] = zero8();
  for (int kk = 0; kk < 128; kk += 32) {
    v16h a = ldA(ab, kk, lane);            // A loaded once, reused 4x
#pragma unroll
    for (int u = 0; u < 4; ++u)
      acc[u] = wmma16(a, ldB(bb[u], kk, lane), acc[u]);
  }
  int mhi = (lane & 16) ? 8 : 0;
  float sqi[8];
#pragma unroll
  for (int r = 0; r < 8; ++r) sqi[r] = sq[b * NPTS + ti * 16 + r + mhi];
  float sqj[4];
#pragma unroll
  for (int u = 0; u < 4; ++u) sqj[u] = sq[b * NPTS + tj4 * 64 + u * 16 + m];
#pragma unroll
  for (int r = 0; r < 8; ++r) {
    size_t rowo = ((size_t)(b * NPTS + ti * 16 + r + mhi)) * NPTS + tj4 * 64 + m;
#pragma unroll
    for (int u = 0; u < 4; ++u)
      D[rowo + u * 16] = sqi[r] + sqj[u] - 2.f * acc[u][r];
  }
}

// ---------- top-(K+1) smallest per row, drop self ----------
__global__ void k_topk(const float* D, int* idx) {
  int t = blockIdx.x * blockDim.x + threadIdx.x;   // B*N
  const float* row = D + (size_t)t * NPTS;
  float bd[KNN + 1]; int bi[KNN + 1];
#pragma unroll
  for (int i = 0; i <= KNN; ++i) { bd[i] = 3.4e38f; bi[i] = -1; }
  for (int j = 0; j < NPTS; ++j) {
    if ((j & 63) == 0 && j + 256 < NPTS)
      __builtin_prefetch(row + j + 256, 0, 3);     // global_prefetch_b8
    float d = row[j];
    if (d < bd[KNN]) {                 // strict: ties keep earlier (lower) index
      int p = KNN;
      while (p > 0 && d < bd[p - 1]) { bd[p] = bd[p - 1]; bi[p] = bi[p - 1]; --p; }
      bd[p] = d; bi[p] = j;
    }
  }
  int* o = idx + (size_t)t * KNN;
  for (int k = 0; k < KNN; ++k) o[k] = bi[k + 1];  // drop self (dist 0)
}

// ---------- edge conv GEMM with LDS-staged gathered operands ----------
// One block = one 16-row tile of flattened (n,k) rows, covering ALL channel tiles:
// (blockDim/32) waves x COT channel tiles each. Gathered neighbor/center rows are
// staged to LDS once and every wave feeds its WMMAs from LDS (ds_load_b128).
// e = W1 @ nb + (W2-W1) @ center.
template <int COT>
__global__ void k_edge_lds(const _Float16* h16, const int* idx,
                           const _Float16* W1h, const _Float16* Wmh,
                           _Float16* e16, float* sum, float* sumsq, int Cout) {
  __shared__ alignas(16) _Float16 s_nb[16 * 128];
  __shared__ alignas(16) _Float16 s_cen[16 * 128];
  int tid = threadIdx.x;
  int lane = tid & 31;
  int wid = tid >> 5;
  int rowT = blockIdx.x & 4095;                    // (N*K)/16 row tiles
  int b = blockIdx.x >> 12;

  // cooperative gather: 16 rows x 128 halves = 256 chunks of 8 halves, per matrix
  for (int i = tid; i < 256; i += blockDim.x) {
    int mr = i >> 4, part = i & 15;                // row, 16B chunk
    int grow = rowT * 16 + mr;
    int n = grow >> 5, kN = grow & 31;
    int j = idx[(size_t)(b * NPTS + n) * KNN + kN];
    *(uint4*)(s_nb + mr * 128 + part * 8) =
        *(const uint4*)(h16 + ((size_t)(b * NPTS + j) << 7) + part * 8);
    *(uint4*)(s_cen + mr * 128 + part * 8) =
        *(const uint4*)(h16 + ((size_t)(b * NPTS + n) << 7) + part * 8);
  }
  __syncthreads();

  int m = lane & 15;
  const _Float16* snb = s_nb + m * 128;
  const _Float16* scen = s_cen + m * 128;
  const _Float16* w1[COT]; const _Float16* wm[COT]; int co[COT];
#pragma unroll
  for (int u = 0; u < COT; ++u) {
    co[u] = ((wid * COT + u) << 4) + m;
    w1[u] = W1h + (size_t)co[u] * 128;
    wm[u] = Wmh + (size_t)co[u] * 128;
  }
  v8f acc[COT];
#pragma unroll
  for (int u = 0; u < COT; ++u) acc[u] = zero8();
  for (int kk = 0; kk < 128; kk += 32) {
    v16h anb  = ldA(snb,  kk, lane);               // LDS reads, reused COT times
    v16h acen = ldA(scen, kk, lane);
#pragma unroll
    for (int u = 0; u < COT; ++u) {
      acc[u] = wmma16(anb,  ldB(w1[u], kk, lane), acc[u]);
      acc[u] = wmma16(acen, ldB(wm[u], kk, lane), acc[u]);
    }
  }
  int mhi = (lane & 16) ? 8 : 0;
#pragma unroll
  for (int u = 0; u < COT; ++u) {
    float s = 0.f, s2 = 0.f;
#pragma unroll
    for (int r = 0; r < 8; ++r) { s += acc[u][r]; s2 += acc[u][r] * acc[u][r]; }
    atomicAdd(&sum[b * Cout + co[u]], s);
    atomicAdd(&sumsq[b * Cout + co[u]], s2);
#pragma unroll
    for (int r = 0; r < 8; ++r) {
      int gr = rowT * 16 + r + mhi;
      e16[((size_t)(b * NPTS * KNN + gr)) * Cout + co[u]] = (_Float16)acc[u][r];
    }
  }
}

// ---------- InstanceNorm2d + LeakyReLU + max over K ----------
__global__ void k_norm_max(const _Float16* e16, const float* sum, const float* sumsq,
                           float* feats, _Float16* h16, int Cout, int featOff, int writeH16) {
  int t = blockIdx.x * blockDim.x + threadIdx.x;   // B*N*Cout
  int co = t & (Cout - 1);
  int n = (t / Cout) & (NPTS - 1);
  int b = t / (Cout * NPTS);
  const float inv = 1.f / (float)(NPTS * KNN);
  float mu = sum[b * Cout + co] * inv;
  float var = sumsq[b * Cout + co] * inv - mu * mu;
  float rs = rsqrtf(var + EPSN);
  const _Float16* e = e16 + ((size_t)((b * NPTS + n) * KNN)) * Cout + co;
  float mx = -3.4e38f;
  for (int k = 0; k < KNN; ++k) {
    float v = (float)e[(size_t)k * Cout];
    v = (v - mu) * rs;
    v = v >= 0.f ? v : SLOPE * v;
    mx = fmaxf(mx, v);
  }
  feats[(size_t)(b * NPTS + n) * 480 + featOff + co] = mx;
  if (writeH16) h16[(size_t)(b * NPTS + n) * 128 + co] = (_Float16)mx;
}

// ---------- final projection 480 -> 128, 16x64 tile/wave ----------
__global__ void k_final4(const _Float16* feats16, const _Float16* wh16,
                         float* f, float* sum, float* sumsq) {
  int lane = threadIdx.x & 31;
  int wave = blockIdx.x * (blockDim.x >> 5) + (threadIdx.x >> 5); // B*128*2 waves
  int coG = wave & 1, nT = (wave >> 1) & 127, b = wave >> 8;
  int m = lane & 15;
  const _Float16* A = feats16 + (size_t)(b * NPTS + nT * 16 + m) * 480;
  const _Float16* Bp[4]; int co[4];
#pragma unroll
  for (int u = 0; u < 4; ++u) {
    co[u] = coG * 64 + u * 16 + m;
    Bp[u] = wh16 + (size_t)co[u] * 480;
  }
  v8f acc[4];
#pragma unroll
  for (int u = 0; u < 4; ++u) acc[u] = zero8();
  for (int kk = 0; kk < 480; kk += 32) {
    v16h a = ldA(A, kk, lane);
#pragma unroll
    for (int u = 0; u < 4; ++u)
      acc[u] = wmma16(a, ldB(Bp[u], kk, lane), acc[u]);
  }
  int mhi = (lane & 16) ? 8 : 0;
#pragma unroll
  for (int u = 0; u < 4; ++u) {
    float s = 0.f, s2 = 0.f;
#pragma unroll
    for (int r = 0; r < 8; ++r) { s += acc[u][r]; s2 += acc[u][r] * acc[u][r]; }
    atomicAdd(&sum[b * 128 + co[u]], s);
    atomicAdd(&sumsq[b * 128 + co[u]], s2);
#pragma unroll
    for (int r = 0; r < 8; ++r)
      f[(size_t)(b * NPTS + nT * 16 + r + mhi) * 128 + co[u]] = acc[u][r];
  }
}

// ---------- InstanceNorm1d + LeakyReLU + global max pool over N ----------
__global__ void k_fnorm(const float* f, const float* sum, const float* sumsq, float* g) {
  int t = blockIdx.x * blockDim.x + threadIdx.x;   // B*128
  int b = t >> 7, co = t & 127;
  const float inv = 1.f / (float)NPTS;
  float mu = sum[t] * inv;
  float var = sumsq[t] * inv - mu * mu;
  float rs = rsqrtf(var + EPSN);
  float mx = -3.4e38f;
  for (int n = 0; n < NPTS; ++n) {
    float v = f[(size_t)(b * NPTS + n) * 128 + co];
    v = (v - mu) * rs;
    v = v >= 0.f ? v : SLOPE * v;
    mx = fmaxf(mx, v);
  }
  g[t] = mx;
}

// ---------- MLP head: fc1 -> rowwise norm -> lrelu -> fc2 ----------
__global__ void k_head(const float* g, const float* fc1_w, const float* fc1_b,
                       const float* fc2_w, const float* fc2_b, float* out) {
  __shared__ float sg[128], sh[128];
  int b = blockIdx.x, t = threadIdx.x;
  sg[t] = g[b * 128 + t];
  __syncthreads();
  float a = fc1_b[t];
  for (int c = 0; c < 128; ++c) a += sg[c] * fc1_w[t * 128 + c];
  sh[t] = a;
  __syncthreads();
  float mu = 0.f;
  for (int c = 0; c < 128; ++c) mu += sh[c];
  mu *= (1.f / 128.f);
  float var = 0.f;
  for (int c = 0; c < 128; ++c) { float d = sh[c] - mu; var += d * d; }
  var *= (1.f / 128.f);
  float v = (a - mu) * rsqrtf(var + EPSN);
  v = v >= 0.f ? v : SLOPE * v;
  __syncthreads();
  sh[t] = v;
  __syncthreads();
  float o = fc2_b[t];
  for (int c = 0; c < 128; ++c) o += sh[c] * fc2_w[t * 128 + c];
  out[b * 128 + t] = o;
}

extern "C" void kernel_launch(void* const* d_in, const int* in_sizes, int n_in,
                              void* d_out, int out_size, void* d_ws, size_t ws_size,
                              hipStream_t stream) {
  const float* x = (const float*)d_in[0];
  // d_in[1] = normals, unused (include_normals=False)
  const float* Ws[4] = {(const float*)d_in[2], (const float*)d_in[3],
                        (const float*)d_in[4], (const float*)d_in[5]};
  const float* Wh    = (const float*)d_in[6];
  const float* fc1_w = (const float*)d_in[7];
  const float* fc1_b = (const float*)d_in[8];
  const float* fc2_w = (const float*)d_in[9];
  const float* fc2_b = (const float*)d_in[10];
  float* out = (float*)d_out;

  char* ws = (char*)d_ws;
  size_t off = 0;
  auto alloc = [&](size_t bytes) -> void* {
    void* p = ws + off;
    off += (bytes + 255) & ~(size_t)255;
    return p;
  };
  _Float16* h16     = (_Float16*)alloc((size_t)BATCH * NPTS * 128 * 2);   // padded features
  _Float16* feats16 = (_Float16*)alloc((size_t)BATCH * NPTS * 480 * 2);
  float*    sq      = (float*)   alloc((size_t)BATCH * NPTS * 4);
  float*    D       = (float*)   alloc((size_t)BATCH * NPTS * NPTS * 4);
  int*      idx     = (int*)     alloc((size_t)BATCH * NPTS * KNN * 4);
  _Float16* W1h     = (_Float16*)alloc((size_t)256 * 128 * 2);
  _Float16* Wmh     = (_Float16*)alloc((size_t)256 * 128 * 2);
  _Float16* Wh16    = (_Float16*)alloc((size_t)128 * 480 * 2);
  float*    stats   = (float*)   alloc((size_t)BATCH * 256 * 2 * 4);      // sum | sumsq
  _Float16* e16     = (_Float16*)alloc((size_t)BATCH * NPTS * KNN * 256 * 2);
  float*    featsF  = (float*)   alloc((size_t)BATCH * NPTS * 480 * 4);
  float*    fbuf    = (float*)   alloc((size_t)BATCH * NPTS * 128 * 4);
  float*    g       = (float*)   alloc((size_t)BATCH * 128 * 4);
  float*    sumsq   = stats + BATCH * 256;

  const int CinA[4]    = {3, 32, 64, 128};
  const int CoutA[4]   = {32, 64, 128, 256};
  const int featOff[4] = {0, 32, 96, 224};

  k_convert_x<<<(BATCH * NPTS * 128) / 256, 256, 0, stream>>>(x, h16);

  for (int L = 0; L < 4; ++L) {
    k_sq<<<(BATCH * NPTS) / 256, 256, 0, stream>>>(h16, sq);
    k_dist4<<<(BATCH * 128 * 32) / 4, 128, 0, stream>>>(h16, sq, D);
    k_topk<<<(BATCH * NPTS) / 128, 128, 0, stream>>>(D, idx);
    k_prep_w<<<(CoutA[L] * 128) / 256, 256, 0, stream>>>(Ws[L], W1h, Wmh, CinA[L]);
    k_zero_f32<<<2, 256, 0, stream>>>(stats, BATCH * 256 * 2);
    // one block per 16-row tile; block covers all channel tiles of the layer
    const int blocks = BATCH * 4096;
    if (L == 0)
      k_edge_lds<1><<<blocks, 64, 0, stream>>>(h16, idx, W1h, Wmh, e16, stats, sumsq,
                                               CoutA[L]);     // 2 waves x 1 tile
    else if (L == 1)
      k_edge_lds<1><<<blocks, 128, 0, stream>>>(h16, idx, W1h, Wmh, e16, stats, sumsq,
                                                CoutA[L]);    // 4 waves x 1 tile
    else if (L == 2)
      k_edge_lds<2><<<blocks, 128, 0, stream>>>(h16, idx, W1h, Wmh, e16, stats, sumsq,
                                                CoutA[L]);    // 4 waves x 2 tiles
    else
      k_edge_lds<4><<<blocks, 128, 0, stream>>>(h16, idx, W1h, Wmh, e16, stats, sumsq,
                                                CoutA[L]);    // 4 waves x 4 tiles
    if (L < 3 && CoutA[L] < 128)
      k_zero_f16<<<(BATCH * NPTS * 128) / 256, 256, 0, stream>>>(h16, BATCH * NPTS * 128);
    k_norm_max<<<(BATCH * NPTS * CoutA[L]) / 256, 256, 0, stream>>>(
        e16, stats, sumsq, featsF, h16, CoutA[L], featOff[L], (L < 3) ? 1 : 0);
  }

  k_cvt_f32_f16<<<(BATCH * NPTS * 480) / 256, 256, 0, stream>>>(featsF, feats16,
                                                                BATCH * NPTS * 480);
  k_cvt_f32_f16<<<(128 * 480) / 256, 256, 0, stream>>>(Wh, Wh16, 128 * 480);
  k_zero_f32<<<2, 256, 0, stream>>>(stats, BATCH * 256 * 2);
  k_final4<<<(BATCH * 128 * 2) / 4, 128, 0, stream>>>(feats16, Wh16, fbuf, stats, sumsq);
  k_fnorm<<<1, 256, 0, stream>>>(fbuf, stats, sumsq, g);
  k_head<<<BATCH, 128, 0, stream>>>(g, fc1_w, fc1_b, fc2_w, fc2_b, out);
}
